// Bottleneck_DSConv_63797444214791
// MI455X (gfx1250) — matrix operand order repaired and checked by reading, not compiled, not run on hardware
//
#include <hip/hip_runtime.h>
#include <hip/hip_bf16.h>
#include <math.h>

// ---------------------------------------------------------------------------
// Types
// ---------------------------------------------------------------------------
typedef __attribute__((ext_vector_type(16))) __bf16 v16bf;
typedef __attribute__((ext_vector_type(8)))  float  v8f;

struct __align__(16) U4 { unsigned int x, y, z, w; };

union Pack16 { unsigned short u[16]; U4 q[2]; };
union Pack8  { unsigned short u[8];  U4 q;    };
union FragBF { U4 q[2]; v16bf v; };

__device__ __forceinline__ float bf2f(unsigned short u) {
    union { unsigned int i; float f; } c; c.i = ((unsigned int)u) << 16; return c.f;
}
__device__ __forceinline__ unsigned short f2bf(float f) {
    union { float f; unsigned int i; } c; c.f = f;
    unsigned int u = c.i;
    u = (u + 0x7fffu + ((u >> 16) & 1u)) >> 16;     // round-to-nearest-even
    return (unsigned short)u;
}
__device__ __forceinline__ float silu_f(float v)  { return v / (1.0f + expf(-v)); }
__device__ __forceinline__ float gelu_f(float v)  { return 0.5f * v * (1.0f + erff(v * 0.70710678118654752f)); }

// ---------------------------------------------------------------------------
// Generic implicit-GEMM convolution with WMMA (bf16 in, f32 accumulate).
// All geometry is compile-time so every div/mod strength-reduces.
//  A (virtual im2col): act is NHWC bf16, k = tap*CIN + ci  (WKIND==0)
//                      act is feat [M][576], k = tap*64 + c (WKIND==1, dsc w (O,C,K))
//  Block: 128 threads = 4 waves; tile 64(M) x 64(N); K-chunk 32.
//  EPI: 0 = +bias -> f32      1 = affine+GELU -> bf16
//       2 = affine+SiLU->bf16 3 = affine+SiLU -> f32
//       4 = tanh((x+b)*g+beta) -> f32
// ---------------------------------------------------------------------------
template<int EPI, int WKIND, int CIN, int COUT, int KTOT, int R, int S, int PAD>
__global__ __launch_bounds__(128)
void conv_gemm_wmma(const unsigned short* __restrict__ act,
                    const float* __restrict__ wgt,
                    const float* __restrict__ p0,   // gain
                    const float* __restrict__ p1,   // bias
                    const float* __restrict__ p2,   // beta (EPI4)
                    void* __restrict__ outv,
                    int ldc, int colOff)
{
    constexpr int H = 80, W = 80, HW = H * W;

    __shared__ unsigned short As[64 * 32];   // [m][k]
    __shared__ unsigned short Bs[64 * 32];   // [n][k] (transposed weights)

    const int tid  = threadIdx.x;
    const int wid  = tid >> 5;
    const int lane = tid & 31;
    const int half = lane >> 4;
    const int l15  = lane & 15;
    const int mBase = blockIdx.x * 64;
    const int nBase = blockIdx.y * 64;

    v8f acc[4];
    #pragma unroll
    for (int i = 0; i < 4; ++i) acc[i] = (v8f)0.0f;

    // staging coords: thread t stages 16 k's of row (t>>1), k0 = (t&1)*16
    const int srow = tid >> 1;
    const int sk0  = (tid & 1) * 16;
    const int am   = mBase + srow;
    const int ab   = am / HW;                 // compile-time divisor
    const int arem = am - ab * HW;
    const int ay   = arem / W;
    const int ax   = arem - ay * W;
    const int bn   = nBase + srow;

    for (int kc = 0; kc < KTOT; kc += 32) {
        // ---- stage A tile (im2col gather, 16 contiguous channels) ----
        {
            const int kg  = kc + sk0;               // CIN mult of 16 -> tap const over run
            const int tap = kg / CIN;
            const int cib = kg - tap * CIN;
            const int ky  = tap / S, kx = tap - ky * S;
            const int yy  = ay + ky - PAD, xx = ax + kx - PAD;
            Pack16 pa;
            if (yy >= 0 && yy < H && xx >= 0 && xx < W) {
                const U4* src = (const U4*)(act + (size_t)((ab * H + yy) * W + xx) * CIN + cib);
                pa.q[0] = src[0]; pa.q[1] = src[1];
            } else {
                pa.q[0] = U4{0,0,0,0}; pa.q[1] = U4{0,0,0,0};
            }
            *(U4*)&As[srow * 32 + sk0]     = pa.q[0];
            *(U4*)&As[srow * 32 + sk0 + 8] = pa.q[1];
        }
        // ---- stage B tile (weight repack fp32 -> bf16, transposed [n][k]) ----
        {
            Pack16 pb;
            #pragma unroll
            for (int j = 0; j < 16; ++j) {
                const int kg = kc + sk0 + j;
                float wv = 0.0f;
                if (bn < COUT) {
                    if (WKIND == 0) {
                        const int tap = kg / CIN, ci = kg - tap * CIN;
                        const int ky = tap / S, kx = tap - ky * S;
                        wv = wgt[(((size_t)bn * CIN + ci) * R + ky) * S + kx];
                    } else { // dsc weights (O, 64, 9), k = tap*64 + c
                        const int tap = kg >> 6, c = kg & 63;
                        wv = wgt[((size_t)bn * 64 + c) * 9 + tap];
                    }
                }
                pb.u[j] = f2bf(wv);
            }
            *(U4*)&Bs[srow * 32 + sk0]     = pb.q[0];
            *(U4*)&Bs[srow * 32 + sk0 + 8] = pb.q[1];
        }
        __syncthreads();

        // ---- A fragment: lanes 0-15 -> K{0-7,16-23}; lanes 16-31 -> K{8-15,24-31}
        FragBF fa;
        {
            const unsigned short* base = &As[(wid * 16 + l15) * 32];
            fa.q[0] = *(const U4*)(base + 8 * half);
            fa.q[1] = *(const U4*)(base + 16 + 8 * half);
        }
        // ---- 4 N-tiles of B, 4 WMMA
        #pragma unroll
        for (int nt = 0; nt < 4; ++nt) {
            FragBF fb;
            const unsigned short* bb = &Bs[(nt * 16 + l15) * 32];
            fb.q[0] = *(const U4*)(bb + 8 * half);
            fb.q[1] = *(const U4*)(bb + 16 + 8 * half);
            acc[nt] = __builtin_amdgcn_wmma_f32_16x16x32_bf16(
                false, fa.v, false, fb.v, (short)0, acc[nt], false, false);
        }
        __syncthreads();
    }

    // ---- epilogue: C/D layout — VGPR r holds M = r + 8*half, N = l15
    #pragma unroll
    for (int nt = 0; nt < 4; ++nt) {
        const int n = nBase + nt * 16 + l15;
        if (n >= COUT) continue;
        float g = 0.f, bia = 0.f, bet = 0.f;
        if (EPI == 1 || EPI == 2 || EPI == 3) { g = p0[n]; bia = p1[n]; }
        else if (EPI == 4)                    { g = p0[n]; bia = p1[n]; bet = p2[n]; }
        else                                  { bia = p1[n]; }
        #pragma unroll
        for (int r = 0; r < 8; ++r) {
            const int m = mBase + wid * 16 + 8 * half + r;
            float v = acc[nt][r];
            const size_t oidx = (size_t)m * ldc + colOff + n;
            if (EPI == 0)      { ((float*)outv)[oidx] = v + bia; }
            else if (EPI == 1) { v = gelu_f(v * g + bia); ((unsigned short*)outv)[oidx] = f2bf(v); }
            else if (EPI == 2) { v = silu_f(v * g + bia); ((unsigned short*)outv)[oidx] = f2bf(v); }
            else if (EPI == 3) { v = silu_f(v * g + bia); ((float*)outv)[oidx] = v; }
            else               { v = tanhf((v + bia) * g + bet); ((float*)outv)[oidx] = v; }
        }
    }
}

// ---------------------------------------------------------------------------
// NCHW f32 -> NHWC bf16  (C=128 fixed)
// ---------------------------------------------------------------------------
__global__ __launch_bounds__(256)
void cvt_x_kernel(const float* __restrict__ x, unsigned short* __restrict__ xn)
{
    int idx = blockIdx.x * blockDim.x + threadIdx.x;
    if (idx >= 25600 * 128) return;
    const int p = idx % 6400;
    const int t = idx / 6400;
    const int c = t & 127;
    const int b = t >> 7;
    xn[(size_t)(b * 6400 + p) * 128 + c] = f2bf(x[idx]);
}

// ---------------------------------------------------------------------------
// Snake bilinear sampling -> feat [M][9*64] bf16 (pre-im2col'd GEMM A)
// off: [M][32] f32 (tanh'd offsets, cols 0..17 valid)
// ---------------------------------------------------------------------------
__global__ __launch_bounds__(256)
void sample_kernel(const unsigned short* __restrict__ h, const float* __restrict__ off,
                   unsigned short* __restrict__ feat, int morph)
{
    int idx = blockIdx.x * blockDim.x + threadIdx.x;
    if (idx >= 25600 * 9) return;
    const int m = idx / 9, k = idx - m * 9;
    const int b = m / 6400, rem = m - b * 6400;
    const int y = rem / 80, x = rem - y * 80;

    const float* ob = off + (size_t)m * 32 + (morph ? 9 : 0);
    float cum = 0.f;
    if (k > 4)      { for (int j = 5; j <= k; ++j) cum += ob[j]; }
    else if (k < 4) { for (int j = k; j <= 3; ++j) cum += ob[j]; }

    float ys, xs;
    if (morph == 0) { ys = (float)y + cum;          xs = (float)x + (float)(k - 4); }
    else            { xs = (float)x + cum;          ys = (float)y + (float)(k - 4); }
    ys = fminf(fmaxf(ys, 0.f), 79.f);
    xs = fminf(fmaxf(xs, 0.f), 79.f);
    const float y0f = floorf(ys), x0f = floorf(xs);
    const float wy = ys - y0f, wx = xs - x0f;
    const int y0 = (int)y0f, x0 = (int)x0f;
    const int y1 = min(y0 + 1, 79), x1 = min(x0 + 1, 79);
    const float w00 = (1.f - wy) * (1.f - wx), w01 = (1.f - wy) * wx;
    const float w10 = wy * (1.f - wx),          w11 = wy * wx;
    const size_t p00 = ((size_t)((b * 80 + y0) * 80 + x0)) * 64;
    const size_t p01 = ((size_t)((b * 80 + y0) * 80 + x1)) * 64;
    const size_t p10 = ((size_t)((b * 80 + y1) * 80 + x0)) * 64;
    const size_t p11 = ((size_t)((b * 80 + y1) * 80 + x1)) * 64;
    unsigned short* dst = feat + (size_t)m * 576 + k * 64;

    #pragma unroll
    for (int c0 = 0; c0 < 64; c0 += 8) {
        Pack8 a, bq, cq, dq, o;
        a.q  = *(const U4*)(h + p00 + c0);
        bq.q = *(const U4*)(h + p01 + c0);
        cq.q = *(const U4*)(h + p10 + c0);
        dq.q = *(const U4*)(h + p11 + c0);
        #pragma unroll
        for (int j = 0; j < 8; ++j) {
            float v = w00 * bf2f(a.u[j]) + w01 * bf2f(bq.u[j])
                    + w10 * bf2f(cq.u[j]) + w11 * bf2f(dq.u[j]);
            o.u[j] = f2bf(v);
        }
        *(U4*)(dst + c0) = o.q;
    }
}

// ---------------------------------------------------------------------------
// GroupNorm stats (32 groups of 4 channels, per batch): block = (b*32+g)
// dsc: [M][128] f32 ; st: [128][2] = {mean, rstd}
// ---------------------------------------------------------------------------
__global__ __launch_bounds__(256)
void gn_stats_kernel(const float* __restrict__ dsc, float* __restrict__ st)
{
    const int bg = blockIdx.x;
    const int b = bg >> 5, g = bg & 31;
    __shared__ float ss[256], sq[256];
    float s = 0.f, q = 0.f;
    for (int i = threadIdx.x; i < 25600; i += 256) {
        const int qq = i / 6400, p = i - qq * 6400;
        const float v = dsc[(size_t)(b * 6400 + p) * 128 + g * 4 + qq];
        s += v; q += v * v;
    }
    ss[threadIdx.x] = s; sq[threadIdx.x] = q; __syncthreads();
    for (int o = 128; o > 0; o >>= 1) {
        if (threadIdx.x < o) { ss[threadIdx.x] += ss[threadIdx.x + o]; sq[threadIdx.x] += sq[threadIdx.x + o]; }
        __syncthreads();
    }
    if (threadIdx.x == 0) {
        const float mu = ss[0] / 25600.f;
        const float var = sq[0] / 25600.f - mu * mu;
        st[bg * 2] = mu; st[bg * 2 + 1] = rsqrtf(var + 1e-5f);
    }
}

__global__ __launch_bounds__(256)
void gn_apply_kernel(const float* __restrict__ dsc, const float* __restrict__ st,
                     const float* __restrict__ gg, const float* __restrict__ gb,
                     unsigned short* __restrict__ ocat, int colOff)
{
    int idx = blockIdx.x * blockDim.x + threadIdx.x;
    if (idx >= 25600 * 128) return;
    const int m = idx >> 7, c = idx & 127;
    const int b = m / 6400, g = c >> 2;
    const float mu = st[(b * 32 + g) * 2], rs = st[(b * 32 + g) * 2 + 1];
    float v = (dsc[idx] - mu) * rs * gg[c] + gb[c];
    v = silu_f(v);
    ocat[(size_t)m * 384 + colOff + c] = f2bf(v);
}

// ---------------------------------------------------------------------------
// EMA attention stats per (b,c): mean + denom
// ---------------------------------------------------------------------------
__global__ __launch_bounds__(256)
void ema_stats_kernel(const float* __restrict__ fo, float* __restrict__ st)
{
    const int bc = blockIdx.x;
    const int b = bc >> 7, c = bc & 127;
    __shared__ float ss[256], sq[256];
    float s = 0.f, q = 0.f;
    for (int p = threadIdx.x; p < 6400; p += 256) {
        const float v = fo[(size_t)(b * 6400 + p) * 128 + c];
        s += v; q += v * v;
    }
    ss[threadIdx.x] = s; sq[threadIdx.x] = q; __syncthreads();
    for (int o = 128; o > 0; o >>= 1) {
        if (threadIdx.x < o) { ss[threadIdx.x] += ss[threadIdx.x + o]; sq[threadIdx.x] += sq[threadIdx.x + o]; }
        __syncthreads();
    }
    if (threadIdx.x == 0) {
        const float sum = ss[0];
        const float mean = sum / 6400.f;
        const float sumd = sq[0] - sum * sum / 6400.f;   // sum (x-mean)^2
        st[bc * 2] = mean;
        st[bc * 2 + 1] = 4.f * (sumd / 6399.f + 1e-4f);
    }
}

__global__ __launch_bounds__(256)
void final_kernel(const float* __restrict__ fo, const float* __restrict__ st,
                  const float* __restrict__ x, float* __restrict__ out)
{
    int idx = blockIdx.x * blockDim.x + threadIdx.x;
    if (idx >= 25600 * 128) return;
    const int m = idx >> 7, c = idx & 127;
    const int b = m / 6400, rem = m - b * 6400;
    const float v = fo[idx];
    const float mean = st[(b * 128 + c) * 2], denom = st[(b * 128 + c) * 2 + 1];
    const float d = v - mean;
    const float e = d * d / denom + 0.5f;
    const float r = v / (1.f + expf(-e));               // v * sigmoid(e)
    const size_t xi = (size_t)(b * 128 + c) * 6400 + rem;
    out[xi] = r + x[xi];
}

// ---------------------------------------------------------------------------
// Host launcher
// ---------------------------------------------------------------------------
extern "C" void kernel_launch(void* const* d_in, const int* in_sizes, int n_in,
                              void* d_out, int out_size, void* d_ws, size_t ws_size,
                              hipStream_t stream)
{
    (void)in_sizes; (void)n_in; (void)out_size; (void)ws_size;
    const float* x        = (const float*)d_in[0];
    const float* cv1_w    = (const float*)d_in[1];
    const float* cv1_g    = (const float*)d_in[2];
    const float* cv1_b    = (const float*)d_in[3];
    const float* conv0_w  = (const float*)d_in[4];
    const float* conv0_g  = (const float*)d_in[5];
    const float* conv0_b  = (const float*)d_in[6];
    const float* off_w[2]    = { (const float*)d_in[7],  (const float*)d_in[15] };
    const float* off_b[2]    = { (const float*)d_in[8],  (const float*)d_in[16] };
    const float* off_g[2]    = { (const float*)d_in[9],  (const float*)d_in[17] };
    const float* off_beta[2] = { (const float*)d_in[10], (const float*)d_in[18] };
    const float* dsc_w[2]    = { (const float*)d_in[11], (const float*)d_in[19] };
    const float* dsc_b[2]    = { (const float*)d_in[12], (const float*)d_in[20] };
    const float* gn_g[2]     = { (const float*)d_in[13], (const float*)d_in[21] };
    const float* gn_b[2]     = { (const float*)d_in[14], (const float*)d_in[22] };
    const float* fuse_w   = (const float*)d_in[23];
    const float* fuse_g   = (const float*)d_in[24];
    const float* fuse_b   = (const float*)d_in[25];

    // workspace carve-up (all 256B aligned); total ~= 88.6 MB
    char* ws = (char*)d_ws;
    size_t cur = 0;
    auto carve = [&](size_t bytes) { void* p = ws + cur; cur += (bytes + 255) & ~(size_t)255; return p; };
    unsigned short* x_nhwc  = (unsigned short*)carve((size_t)25600 * 128 * 2);
    unsigned short* h_nhwc  = (unsigned short*)carve((size_t)25600 * 64 * 2);
    float*          offbuf  = (float*)         carve((size_t)25600 * 32 * 4);
    unsigned short* feat    = (unsigned short*)carve((size_t)25600 * 576 * 2);
    float*          dsc_out = (float*)         carve((size_t)25600 * 128 * 4);
    unsigned short* o_cat   = (unsigned short*)carve((size_t)25600 * 384 * 2);
    float*          fuse_o  = (float*)         carve((size_t)25600 * 128 * 4);
    float*          gnstats = (float*)         carve((size_t)128 * 2 * 4);
    float*          emast   = (float*)         carve((size_t)512 * 2 * 4);

    const dim3 blk128(128), blk256(256);

    // 1) x (NCHW f32) -> NHWC bf16
    cvt_x_kernel<<<(25600 * 128 + 255) / 256, blk256, 0, stream>>>(x, x_nhwc);

    // 2) cv1: 3x3, 128->64, affine+GELU -> h_nhwc bf16
    conv_gemm_wmma<1, 0, 128, 64, 1152, 3, 3, 1><<<dim3(400, 1), blk128, 0, stream>>>(
        x_nhwc, cv1_w, cv1_g, cv1_b, nullptr, h_nhwc, 64, 0);

    // 3) conv0: 9x9, 64->128, affine+SiLU -> o_cat cols [0,128) bf16
    conv_gemm_wmma<2, 0, 64, 128, 5184, 9, 9, 4><<<dim3(400, 2), blk128, 0, stream>>>(
        h_nhwc, conv0_w, conv0_g, conv0_b, nullptr, o_cat, 384, 0);

    // 4/5) snake branches (morph 0 = ox, morph 1 = oy); reuse offbuf/feat/dsc_out
    for (int br = 0; br < 2; ++br) {
        // offset conv: 3x3, 64->18, tanh((conv+b)*g+beta) -> offbuf [M][32] f32
        conv_gemm_wmma<4, 0, 64, 18, 576, 3, 3, 1><<<dim3(400, 1), blk128, 0, stream>>>(
            h_nhwc, off_w[br], off_g[br], off_b[br], off_beta[br], offbuf, 32, 0);
        // bilinear snake sampling -> feat [M][576] bf16
        sample_kernel<<<(25600 * 9 + 255) / 256, blk256, 0, stream>>>(h_nhwc, offbuf, feat, br);
        // snake einsum: GEMM M=25600, N=128, K=576, +bias -> dsc_out f32
        conv_gemm_wmma<0, 1, 576, 128, 576, 1, 1, 0><<<dim3(400, 2), blk128, 0, stream>>>(
            feat, dsc_w[br], nullptr, dsc_b[br], nullptr, dsc_out, 128, 0);
        // GroupNorm(32 groups) + SiLU -> o_cat cols [128+128*br, ...)
        gn_stats_kernel<<<128, blk256, 0, stream>>>(dsc_out, gnstats);
        gn_apply_kernel<<<(25600 * 128 + 255) / 256, blk256, 0, stream>>>(
            dsc_out, gnstats, gn_g[br], gn_b[br], o_cat, 128 + 128 * br);
    }

    // 6) fuse: 1x1, 384->128, affine+SiLU -> fuse_o f32
    conv_gemm_wmma<3, 0, 384, 128, 384, 1, 1, 0><<<dim3(400, 2), blk128, 0, stream>>>(
        o_cat, fuse_w, fuse_g, fuse_b, nullptr, fuse_o, 128, 0);

    // 7) EMA-style spatial attention + residual -> d_out (NCHW f32)
    ema_stats_kernel<<<512, blk256, 0, stream>>>(fuse_o, emast);
    final_kernel<<<(25600 * 128 + 255) / 256, blk256, 0, stream>>>(
        fuse_o, emast, x, (float*)d_out);
}
